// FNS_88304527606524
// MI455X (gfx1250) — matrix operand order repaired and checked by reading, not compiled, not run on hardware
//
#include <hip/hip_runtime.h>
#include <hip/hip_bf16.h>

typedef __attribute__((ext_vector_type(16))) __bf16 v16bf;
typedef __attribute__((ext_vector_type(8)))  float  v8f;

__device__ __forceinline__ unsigned short f2bf(float f) {
    unsigned u = __float_as_uint(f);
    return (unsigned short)((u + 0x7fffu + ((u >> 16) & 1u)) >> 16); // RNE
}
__device__ __forceinline__ float bf2f(unsigned short u) {
    return __uint_as_float(((unsigned)u) << 16);
}
// activation layout: [b][comp(re/im)][y][x][16ch]  (bf16, channel-interleaved)
__device__ __forceinline__ size_t actIdx(int b, int comp, int y, int x) {
    return ((((size_t)b * 2 + comp) * 256 + y) * 256 + x) * 16;
}

// ---------------------------------------------------------------------------
// Odd-symmetric expansion 127x127 -> 256x256 complex (imag = 0)
// ---------------------------------------------------------------------------
__global__ void expand_kernel(const float* __restrict__ r, float2* __restrict__ rsym) {
    unsigned idx = blockIdx.x * 256 + threadIdx.x;      // 32*256*256
    int x = idx & 255, y = (idx >> 8) & 255, b = idx >> 16;
    float v = 0.f;
    if (y != 0 && y != 128 && x != 0 && x != 128) {
        int ry, rx; float s = 1.f;
        if (y <= 127) ry = y - 1; else { ry = 255 - y; s = -s; }
        if (x <= 127) rx = x - 1; else { rx = 255 - x; s = -s; }
        v = s * r[((size_t)b * 127 + ry) * 127 + rx];
    }
    rsym[idx] = make_float2(v, 0.f);
}

// ---------------------------------------------------------------------------
// 256-point radix-2 FFT per line (rows or cols), shift folded as index^128.
// sign=+1 inverse, -1 forward. 128 threads / line.
// ---------------------------------------------------------------------------
__global__ __launch_bounds__(128)
void fft256_kernel(const float2* __restrict__ in, float2* __restrict__ out,
                   int estride, int lstride, float sign, float scale,
                   int shiftIn, int shiftOut) {
    int b = blockIdx.x >> 8, line = blockIdx.x & 255, tid = threadIdx.x;
    __shared__ float re[256], im[256];
    size_t base = (size_t)b * 65536 + (size_t)line * lstride;
    for (int j = tid; j < 256; j += 128) {
        int src = shiftIn ? (j ^ 128) : j;
        float2 v = in[base + (size_t)src * estride];
        int pos = __brev((unsigned)j) >> 24;             // bit-reversed placement
        re[pos] = v.x; im[pos] = v.y;
    }
    __syncthreads();
    for (int len = 2; len <= 256; len <<= 1) {
        int half = len >> 1;
        int g = tid / half, p = tid - g * half;
        int i1 = g * len + p, i2 = i1 + half;
        float ang = sign * 6.28318530717958647f * (float)p / (float)len;
        float s, c; __sincosf(ang, &s, &c);
        float xr = re[i2], xi = im[i2];
        float tr = c * xr - s * xi, ti = c * xi + s * xr;
        float ur = re[i1], ui = im[i1];
        re[i1] = ur + tr; im[i1] = ui + ti;
        re[i2] = ur - tr; im[i2] = ui - ti;
        __syncthreads();
    }
    for (int j = tid; j < 256; j += 128) {
        int dst = shiftOut ? (j ^ 128) : j;
        out[base + (size_t)dst * estride] = make_float2(re[j] * scale, im[j] * scale);
    }
}

// ---------------------------------------------------------------------------
// Stage 1: complex conv 1 -> 16 channels (VALU; 1/16 of main-stage FLOPs)
// ---------------------------------------------------------------------------
__global__ __launch_bounds__(256)
void conv_in_kernel(const float2* __restrict__ fin, unsigned short* __restrict__ out,
                    const float* __restrict__ w1r, const float* __restrict__ w1i) {
    int b = blockIdx.x, y = blockIdx.y, x = threadIdx.x;
    __shared__ float swr[144], swi[144];                 // [co][ky*3+kx]
    if (threadIdx.x < 144) {
        swr[threadIdx.x] = w1r[(size_t)b * 144 + threadIdx.x];
        swi[threadIdx.x] = w1i[(size_t)b * 144 + threadIdx.x];
    }
    __syncthreads();
    float xr[9], xi[9];
#pragma unroll
    for (int t = 0; t < 9; ++t) {
        int yy = y + t / 3 - 1, xx = x + t % 3 - 1;
        if ((unsigned)yy < 256u && (unsigned)xx < 256u) {
            float2 v = fin[((size_t)b * 256 + yy) * 256 + xx];
            xr[t] = v.x; xi[t] = v.y;
        } else { xr[t] = 0.f; xi[t] = 0.f; }
    }
    unsigned short outr[16], outi[16];
#pragma unroll
    for (int co = 0; co < 16; ++co) {
        float ar = 0.f, ai = 0.f;
#pragma unroll
        for (int t = 0; t < 9; ++t) {
            float wr = swr[co * 9 + t], wi = swi[co * 9 + t];
            ar += wr * xr[t] - wi * xi[t];
            ai += wr * xi[t] + wi * xr[t];
        }
        outr[co] = f2bf(ar); outi[co] = f2bf(ai);
    }
    *reinterpret_cast<uint4*>(out + actIdx(b, 0, y, x))     = *reinterpret_cast<uint4*>(&outr[0]);
    *reinterpret_cast<uint4*>(out + actIdx(b, 0, y, x) + 8) = *reinterpret_cast<uint4*>(&outr[8]);
    *reinterpret_cast<uint4*>(out + actIdx(b, 1, y, x))     = *reinterpret_cast<uint4*>(&outi[0]);
    *reinterpret_cast<uint4*>(out + actIdx(b, 1, y, x) + 8) = *reinterpret_cast<uint4*>(&outi[8]);
}

// ---------------------------------------------------------------------------
// Weight A-fragment prep: pack Wr / Wi / -Wi into 16-bit
// A-matrix (16x32) regs, contiguous k = 16*(lane/16)+e, per batch & stage; K = tap*16 + ci,
// taps padded 9 -> 10 (chunk K=32 -> 5 chunks).
// frag layout: [b][stage][variant(3)][chunk(5)][lane(32)][16] bf16
// ---------------------------------------------------------------------------
__global__ void prep_frag_kernel(const float* __restrict__ wr, const float* __restrict__ wi,
                                 unsigned short* __restrict__ frag, int stage, int adjoint) {
    int b = blockIdx.x, lane = threadIdx.x;
    int m = lane & 15, hi = lane >> 4;
    const float* WR = wr + (size_t)b * 16 * 16 * 9;
    const float* WI = wi + (size_t)b * 16 * 16 * 9;
    unsigned short* fb = frag + (((size_t)b * 4 + stage) * 3) * 5 * 32 * 16;
    for (int c = 0; c < 5; ++c) {
        unsigned short vr[16], vi[16], vni[16];
        for (int v = 0; v < 8; ++v) {
            int klo = 16 * hi + 2 * v;
            for (int h = 0; h < 2; ++h) {
                int k = 32 * c + klo + h;
                int tap = k >> 4, ci = k & 15;
                float fr = 0.f, fi = 0.f;
                if (tap < 9) {
                    int ky = tap / 3, kx = tap % 3;
                    if (!adjoint) {
                        fr = WR[((size_t)m * 16 + ci) * 9 + ky * 3 + kx];
                        fi = WI[((size_t)m * 16 + ci) * 9 + ky * 3 + kx];
                    } else {   // adj: swap cin/cout, transpose taps, conjugate
                        fr =  WR[((size_t)ci * 16 + m) * 9 + kx * 3 + ky];
                        fi = -WI[((size_t)ci * 16 + m) * 9 + kx * 3 + ky];
                    }
                }
                vr[2 * v + h] = f2bf(fr); vi[2 * v + h] = f2bf(fi); vni[2 * v + h] = f2bf(-fi);
            }
        }
        unsigned short* d0 = fb + (((size_t)0 * 5 + c) * 32 + lane) * 16;
        unsigned short* d1 = fb + (((size_t)1 * 5 + c) * 32 + lane) * 16;
        unsigned short* d2 = fb + (((size_t)2 * 5 + c) * 32 + lane) * 16;
        for (int e = 0; e < 16; ++e) { d0[e] = vr[e]; d1[e] = vi[e]; d2[e] = vni[e]; }
    }
}

// ---------------------------------------------------------------------------
// Main 16->16 complex conv via WMMA bf16 implicit GEMM.
// Block: 256 thr (8 waves). Tile: 16 rows x 32 cols out; wave-tile 1x16.
// LDS halo tile filled with GLOBAL_LOAD_ASYNC_TO_LDS_B128 (ASYNCcnt path,
// no VGPR round-trip); B fragments are contiguous 32B LDS reads per lane,
// software-pipelined one K-chunk ahead of the 20 v_wmma per tile.
// ---------------------------------------------------------------------------
__global__ __launch_bounds__(256)
void conv16_wmma_kernel(const unsigned short* __restrict__ in,
                        unsigned short* __restrict__ out,
                        const unsigned short* __restrict__ wfrag, int stage) {
    const int b = blockIdx.x;
    const int gy0 = blockIdx.y * 16;
    const int gx0 = blockIdx.z * 32;
    const int tid = threadIdx.x;
    const int lane = tid & 31;
    const int wave = tid >> 5;
    const int n = lane & 15;
    const int hi16 = lane >> 4;

    __shared__ __align__(32) unsigned short sh[2 * 18 * 34 * 16 + 16];
    const int zeroOff = 2 * 18 * 34 * 16;
    if (tid < 16) sh[zeroOff + tid] = 0;

    // stage LDS tile: 2 comps x 18 rows x 34 cols x 16 ch (32B slots)
    // in-bounds slots: async direct-to-LDS copy; OOB halo: zero via ds_store
    for (int s = tid; s < 2 * 18 * 34; s += 256) {
        int comp = s / (18 * 34);
        int rc = s - comp * (18 * 34);
        int rrow = rc / 34, ccol = rc - rrow * 34;
        int gy = gy0 - 1 + rrow, gx = gx0 - 1 + ccol;
        unsigned ldsByte = (unsigned)(unsigned long long)(uintptr_t)&sh[(size_t)s * 16];
        if ((unsigned)gy < 256u && (unsigned)gx < 256u) {
            const unsigned short* gp = in + actIdx(b, comp, gy, gx);
            asm volatile("global_load_async_to_lds_b128 %0, %1, off"
                         :: "v"(ldsByte), "v"(gp) : "memory");
            asm volatile("global_load_async_to_lds_b128 %0, %1, off offset:16"
                         :: "v"(ldsByte), "v"(gp) : "memory");
        } else {
            uint4* d = reinterpret_cast<uint4*>(&sh[(size_t)s * 16]);
            d[0] = make_uint4(0, 0, 0, 0);
            d[1] = make_uint4(0, 0, 0, 0);
        }
    }
    asm volatile("s_wait_asynccnt 0x0" ::: "memory");
    __syncthreads();

    // weight A-fragments (registers): Wr, Wi, -Wi, 5 chunks each
    const unsigned short* wf = wfrag + (((size_t)b * 4 + stage) * 3) * 5 * 32 * 16;
    v16bf aWr[5], aWi[5], aWni[5];
#pragma unroll
    for (int c = 0; c < 5; ++c) {
        aWr[c]  = *reinterpret_cast<const v16bf*>(wf + (((size_t)0 * 5 + c) * 32 + lane) * 16);
        aWi[c]  = *reinterpret_cast<const v16bf*>(wf + (((size_t)1 * 5 + c) * 32 + lane) * 16);
        aWni[c] = *reinterpret_cast<const v16bf*>(wf + (((size_t)2 * 5 + c) * 32 + lane) * 16);
    }

    for (int t = 0; t < 4; ++t) {
        int tile = wave * 4 + t;
        int row = tile >> 1;
        int lx0 = (tile & 1) * 16;
        // B-fragment LDS offsets per K-chunk (tap pair); tap 9 -> zero pad slot
        int offR[5], offI[5];
#pragma unroll
        for (int c = 0; c < 5; ++c) {
            int tap = 2 * c + hi16;                  // lanes 0-15: K=0-15, 16-31: K=16-31
            int dy = tap / 3, dx = tap - dy * 3;
            int base = ((row + dy) * 34 + (lx0 + n + dx)) * 16;
            offR[c] = (tap > 8) ? zeroOff : base;
            offI[c] = (tap > 8) ? zeroOff : (18 * 34 * 16 + base);
        }
        v8f accR = {0.f, 0.f, 0.f, 0.f, 0.f, 0.f, 0.f, 0.f};
        v8f accI = {0.f, 0.f, 0.f, 0.f, 0.f, 0.f, 0.f, 0.f};
        v16bf Br = *reinterpret_cast<const v16bf*>(&sh[offR[0]]);
        v16bf Bi = *reinterpret_cast<const v16bf*>(&sh[offI[0]]);
#pragma unroll
        for (int c = 0; c < 5; ++c) {
            v16bf BrN, BiN;
            if (c < 4) {                              // prefetch next chunk's fragments
                BrN = *reinterpret_cast<const v16bf*>(&sh[offR[c + 1]]);
                BiN = *reinterpret_cast<const v16bf*>(&sh[offI[c + 1]]);
            }
            accR = __builtin_amdgcn_wmma_f32_16x16x32_bf16(false, aWr[c],  false, Br, (short)0, accR, false, false);
            accI = __builtin_amdgcn_wmma_f32_16x16x32_bf16(false, aWi[c],  false, Br, (short)0, accI, false, false);
            accR = __builtin_amdgcn_wmma_f32_16x16x32_bf16(false, aWni[c], false, Bi, (short)0, accR, false, false);
            accI = __builtin_amdgcn_wmma_f32_16x16x32_bf16(false, aWr[c],  false, Bi, (short)0, accI, false, false);
            if (c < 4) { Br = BrN; Bi = BiN; }
        }
        // D layout: lane<16 -> ch 0..7, lane>=16 -> ch 8..15 at pixel col n
        int cibase = hi16 * 8;
        int gy = gy0 + row, gx = gx0 + lx0 + n;
        uint4 pr, pi;
        pr.x = (unsigned)f2bf(accR[0]) | ((unsigned)f2bf(accR[1]) << 16);
        pr.y = (unsigned)f2bf(accR[2]) | ((unsigned)f2bf(accR[3]) << 16);
        pr.z = (unsigned)f2bf(accR[4]) | ((unsigned)f2bf(accR[5]) << 16);
        pr.w = (unsigned)f2bf(accR[6]) | ((unsigned)f2bf(accR[7]) << 16);
        pi.x = (unsigned)f2bf(accI[0]) | ((unsigned)f2bf(accI[1]) << 16);
        pi.y = (unsigned)f2bf(accI[2]) | ((unsigned)f2bf(accI[3]) << 16);
        pi.z = (unsigned)f2bf(accI[4]) | ((unsigned)f2bf(accI[5]) << 16);
        pi.w = (unsigned)f2bf(accI[6]) | ((unsigned)f2bf(accI[7]) << 16);
        *reinterpret_cast<uint4*>(out + actIdx(b, 0, gy, gx) + cibase) = pr;
        *reinterpret_cast<uint4*>(out + actIdx(b, 1, gy, gx) + cibase) = pi;
    }
}

// ---------------------------------------------------------------------------
// h *= wtheta  (elementwise complex; thread layout matches wtheta [b][ch][y][x])
// ---------------------------------------------------------------------------
__global__ void theta_mul_kernel(unsigned short* __restrict__ h,
                                 const float* __restrict__ wtr,
                                 const float* __restrict__ wti) {
    unsigned idx = blockIdx.x * 256 + threadIdx.x;      // 32*16*256*256
    int x = idx & 255, y = (idx >> 8) & 255, ci = (idx >> 16) & 15, b = idx >> 20;
    float tr = wtr[idx], ti = wti[idx];
    size_t iR = actIdx(b, 0, y, x) + ci, iI = actIdx(b, 1, y, x) + ci;
    float hr = bf2f(h[iR]), hi = bf2f(h[iI]);
    h[iR] = f2bf(hr * tr - hi * ti);
    h[iI] = f2bf(hr * ti + hi * tr);
}

// ---------------------------------------------------------------------------
// Final stage: complex conv 16 -> 1 with adj(w1)
// ---------------------------------------------------------------------------
__global__ __launch_bounds__(256)
void conv_out_kernel(const unsigned short* __restrict__ in, float2* __restrict__ fout,
                     const float* __restrict__ w1r, const float* __restrict__ w1i) {
    int b = blockIdx.x, y = blockIdx.y, x = threadIdx.x;
    __shared__ float swr[144], swi[144];                 // adj: transpose taps, conj
    if (threadIdx.x < 144) {
        int ci = threadIdx.x / 9, t = threadIdx.x % 9;
        int ky = t / 3, kx = t % 3;
        swr[ci * 9 + t] =  w1r[(size_t)b * 144 + ci * 9 + kx * 3 + ky];
        swi[ci * 9 + t] = -w1i[(size_t)b * 144 + ci * 9 + kx * 3 + ky];
    }
    __syncthreads();
    float ar = 0.f, ai = 0.f;
#pragma unroll
    for (int t = 0; t < 9; ++t) {
        int yy = y + t / 3 - 1, xx = x + t % 3 - 1;
        if ((unsigned)yy >= 256u || (unsigned)xx >= 256u) continue;
        const unsigned short* pr = in + actIdx(b, 0, yy, xx);
        const unsigned short* pi = in + actIdx(b, 1, yy, xx);
#pragma unroll
        for (int ci = 0; ci < 16; ++ci) {
            float xrv = bf2f(pr[ci]), xiv = bf2f(pi[ci]);
            float wr = swr[ci * 9 + t], wi = swi[ci * 9 + t];
            ar += wr * xrv - wi * xiv;
            ai += wr * xiv + wi * xrv;
        }
    }
    fout[((size_t)b * 256 + y) * 256 + x] = make_float2(ar, ai);
}

__global__ void crop_kernel(const float2* __restrict__ f, float* __restrict__ out, int total) {
    int idx = blockIdx.x * 256 + threadIdx.x;
    if (idx >= total) return;
    int x = idx % 127, y = (idx / 127) % 127, b = idx / (127 * 127);
    out[idx] = f[((size_t)b * 256 + y) * 256 + x].x;
}

// ---------------------------------------------------------------------------
extern "C" void kernel_launch(void* const* d_in, const int* in_sizes, int n_in,
                              void* d_out, int out_size, void* d_ws, size_t ws_size,
                              hipStream_t stream) {
    const float* r   = (const float*)d_in[0];
    const float* w1r = (const float*)d_in[1];
    const float* w1i = (const float*)d_in[2];
    const float* w2r = (const float*)d_in[3];
    const float* w2i = (const float*)d_in[4];
    const float* w3r = (const float*)d_in[5];
    const float* w3i = (const float*)d_in[6];
    const float* wtr = (const float*)d_in[7];
    const float* wti = (const float*)d_in[8];

    char* ws = (char*)d_ws;
    const size_t fftBytes = 32ull * 65536 * sizeof(float2);       // 16.78 MB
    const size_t actBytes = 32ull * 2 * 65536 * 16 * 2;           // 134.2 MB
    float2*          fftA  = (float2*)ws;
    unsigned short*  act0  = (unsigned short*)(ws + fftBytes);
    unsigned short*  act1  = (unsigned short*)(ws + fftBytes + actBytes);
    float2*          fftB  = (float2*)act1;    // alias: disjoint lifetimes
    unsigned short*  wfrag = (unsigned short*)(ws + fftBytes + 2 * actBytes);

    // weight A-fragment prep (stages: w2, w3, adj(w3), adj(w2))
    prep_frag_kernel<<<32, 32, 0, stream>>>(w2r, w2i, wfrag, 0, 0);
    prep_frag_kernel<<<32, 32, 0, stream>>>(w3r, w3i, wfrag, 1, 0);
    prep_frag_kernel<<<32, 32, 0, stream>>>(w3r, w3i, wfrag, 2, 1);
    prep_frag_kernel<<<32, 32, 0, stream>>>(w2r, w2i, wfrag, 3, 1);

    // r_hat = fftshift(ifft2(expand(r)))
    expand_kernel<<<(32 * 65536) / 256, 256, 0, stream>>>(r, fftA);
    fft256_kernel<<<32 * 256, 128, 0, stream>>>(fftA, fftB, 1, 256, +1.f, 1.f, 0, 1);
    fft256_kernel<<<32 * 256, 128, 0, stream>>>(fftB, fftA, 256, 1, +1.f, 1.f / 65536.f, 0, 1);

    // conv chain
    conv_in_kernel<<<dim3(32, 256), 256, 0, stream>>>(fftA, act0, w1r, w1i);
    conv16_wmma_kernel<<<dim3(32, 16, 8), 256, 0, stream>>>(act0, act1, wfrag, 0);
    conv16_wmma_kernel<<<dim3(32, 16, 8), 256, 0, stream>>>(act1, act0, wfrag, 1);
    theta_mul_kernel<<<(32u * 16 * 65536) / 256, 256, 0, stream>>>(act0, wtr, wti);
    conv16_wmma_kernel<<<dim3(32, 16, 8), 256, 0, stream>>>(act0, act1, wfrag, 2);
    conv16_wmma_kernel<<<dim3(32, 16, 8), 256, 0, stream>>>(act1, act0, wfrag, 3);
    conv_out_kernel<<<dim3(32, 256), 256, 0, stream>>>(act0, fftA, w1r, w1i);

    // e = fft2(ifftshift(h)).real, crop
    fft256_kernel<<<32 * 256, 128, 0, stream>>>(fftA, fftB, 1, 256, -1.f, 1.f, 1, 0);
    fft256_kernel<<<32 * 256, 128, 0, stream>>>(fftB, fftA, 256, 1, -1.f, 1.f, 1, 0);
    crop_kernel<<<(32 * 127 * 127 + 255) / 256, 256, 0, stream>>>(fftA, (float*)d_out, 32 * 127 * 127);
}
